// EfficientGraphAttentionBlock_9216999817863
// MI455X (gfx1250) — matrix operand order repaired and checked by reading, not compiled, not run on hardware
//
#include <hip/hip_runtime.h>

// Problem constants (from reference)
#define N_NODES 8192
#define KNB 32
#define HID 256
#define NHEAD 8
#define EXPD 128

typedef __attribute__((ext_vector_type(16))) __bf16 v16bf;
typedef __attribute__((ext_vector_type(8)))  __bf16 v8bf;
typedef __attribute__((ext_vector_type(8)))  float  v8f;

// ---------- bf16 helpers (bit-level, RNE) ----------
__device__ __forceinline__ __bf16 f2bf(float f) {
  unsigned u = __builtin_bit_cast(unsigned, f);
  unsigned r = u + 0x7FFFu + ((u >> 16) & 1u);
  unsigned short hs = (unsigned short)(r >> 16);
  return __builtin_bit_cast(__bf16, hs);
}
__device__ __forceinline__ float bf2f(__bf16 h) {
  unsigned short hs = __builtin_bit_cast(unsigned short, h);
  unsigned u = ((unsigned)hs) << 16;
  return __builtin_bit_cast(float, u);
}
__device__ __forceinline__ v16bf cat8(v8bf lo, v8bf hi) {
  return __builtin_shufflevector(lo, hi, 0,1,2,3,4,5,6,7,8,9,10,11,12,13,14,15);
}

// ---------- WMMA fragment loaders ----------
// A (16x32 bf16, MxK): lane -> m = m0+(lane&15); element e holds
// k = (e>>3)*16 + (lane>>4)*8 + (e&7)  => two contiguous 16B chunks.
__device__ __forceinline__ v16bf load_A_frag(const __bf16* base, int m0, int k0,
                                             int ld, int lane) {
  int m   = m0 + (lane & 15);
  int hlf = lane >> 4;
  const __bf16* p = base + (size_t)m * ld + k0 + hlf * 8;
  v8bf lo = *(const v8bf*)(p);
  v8bf hi = *(const v8bf*)(p + 16);
  return cat8(lo, hi);
}
// B (32x16 bf16, KxN) sourced from a row-major "B-transposed" buffer Bt[n][k]:
// lane -> n = n0+(lane&15); element e holds k = (e>>3)*16 + (lane>>4)*8 + (e&7)  => two contiguous 16B chunks.
__device__ __forceinline__ v16bf load_B_frag(const __bf16* bT, int n0, int k0,
                                             int ld, int lane) {
  int n   = n0 + (lane & 15);
  int hlf = lane >> 4;
  const __bf16* p = bT + (size_t)n * ld + k0 + hlf * 8;
  v8bf lo = *(const v8bf*)(p);
  v8bf hi = *(const v8bf*)(p + 16);
  return cat8(lo, hi);
}
__device__ __forceinline__ void wmma_bf16(v8f& c, v16bf a, v16bf b) {
  c = __builtin_amdgcn_wmma_f32_16x16x32_bf16(false, a, false, b, (short)0, c,
                                              false, false);
}

// One wave computes a 32x32 output block (2x2 tiles of 16x16), K-loop by 32.
__device__ __forceinline__ void gemm_32tile(const __bf16* A, int ldA,
                                            const __bf16* BT, int ldB, int Kdim,
                                            int n0, int lane, v8f (&c)[2][2]) {
  for (int k0 = 0; k0 < Kdim; k0 += 32) {
    v16bf a0 = load_A_frag(A, 0,  k0, ldA, lane);
    v16bf a1 = load_A_frag(A, 16, k0, ldA, lane);
    v16bf b0 = load_B_frag(BT, n0,      k0, ldB, lane);
    v16bf b1 = load_B_frag(BT, n0 + 16, k0, ldB, lane);
    wmma_bf16(c[0][0], a0, b0);
    wmma_bf16(c[0][1], a0, b1);
    wmma_bf16(c[1][0], a1, b0);
    wmma_bf16(c[1][1], a1, b1);
  }
}

struct Params {
  const float* node_features;
  const float* edge_features;
  const float* edx;
  const float* attn_mask;
  const float* angle;
  const float* attn_scalar;
  const __bf16 *WeT, *WnT, *WehT, *WgT, *WcT, *WqT, *WkT, *WvT, *WoT;
  const float *We_b, *Wn_b, *Weh_b, *Wg_b, *Wc_b, *Wq_b, *Wk_b, *Wv_b, *Wo_b;
  const float *ln_e_s, *ln_e_b, *ln_m_s, *ln_m_b;
  const int* nlist;
  const unsigned char* nmask;
  float* outN;
  float* outE;
};

// Row-wise LayerNorm over 256 cols of the f32 scratch [32][260] -> bf16 [32][264].
__device__ __forceinline__ void layernorm_rows(const float* X, __bf16* Y,
                                               const float* sg, const float* bg,
                                               float* red, float* stats, int tid) {
  int r = tid >> 3, seg = tid & 7;
  const float* x = X + r * 260 + seg * 32;
  float s = 0.f, q = 0.f;
#pragma unroll
  for (int i = 0; i < 32; ++i) { float v = x[i]; s += v; q += v * v; }
  red[r * 16 + seg]     = s;
  red[r * 16 + 8 + seg] = q;
  __syncthreads();
  if (tid < 32) {
    float S = 0.f, Q = 0.f;
#pragma unroll
    for (int i = 0; i < 8; ++i) { S += red[tid * 16 + i]; Q += red[tid * 16 + 8 + i]; }
    float mu  = S * (1.f / 256.f);
    float var = Q * (1.f / 256.f) - mu * mu;
    stats[tid * 2]     = mu;
    stats[tid * 2 + 1] = rsqrtf(var + 1e-5f);
  }
  __syncthreads();
  float mu = stats[r * 2], rs = stats[r * 2 + 1];
#pragma unroll
  for (int i = 0; i < 32; ++i) {
    int cc = seg * 32 + i;
    float v = (X[r * 260 + cc] - mu) * rs * sg[cc] + bg[cc];
    Y[r * 264 + cc] = f2bf(v);
  }
  __syncthreads();
}

// Weight convert+transpose: dst[n*Kin + k] = bf16(src[k*Nout + n])
__global__ void wconv_kernel(const float* __restrict__ src, __bf16* __restrict__ dst,
                             int Kin, int Nout) {
  int i = blockIdx.x * blockDim.x + threadIdx.x;
  if (i < Kin * Nout) {
    int n = i / Kin;
    int k = i - n * Kin;
    dst[i] = f2bf(src[(size_t)k * Nout + n]);
  }
}

// One workgroup (256 threads = 8 waves) fully processes one node.
__global__ __launch_bounds__(256)
void ega_fused_kernel(Params p) {
  const int node = blockIdx.x;
  const int tid  = threadIdx.x;
  const int lane = tid & 31;
  const int wave = tid >> 5;
  const int hlf  = lane >> 4;
  const int nl16 = lane & 15;
  const int n0   = wave * 32;  // output-column block owned by this wave

  // ---- LDS pool with phase-based overlays (total 159232 B + small reducers) ----
  __shared__ alignas(16) unsigned char smem[159232];
  __shared__ float sRed[32 * 16];
  __shared__ float sStats[64];

  float*  sScr = (float*) (smem + 0);        // [32][260] f32 scratch
  float*  sSC  = (float*) (smem + 0);        // [8][32][33] attention scores (overlay)
  __bf16* sNF  = (__bf16*)(smem + 33792);    // [32][264] gathered node feats -> later Q
  __bf16* sQ   = sNF;
  __bf16* sEF  = (__bf16*)(smem + 50688);    // [32][264] edge feats -> later K
  __bf16* sKk  = sEF;
  __bf16* sEA  = (__bf16*)(smem + 67584);    // [32][264] ea -> later O
  __bf16* sO   = sEA;
  __bf16* sNH  = (__bf16*)(smem + 84480);    // [32][264]
  __bf16* sEH  = (__bf16*)(smem + 101376);   // [32][264]
  __bf16* sMSG = (__bf16*)(smem + 118272);   // [32][264] -> later attn weights [8][32][40]
  __bf16* sAT  = sMSG;
  __bf16* sEDX = (__bf16*)(smem + 138752);   // [32][136] -> later V^T [256][40]
  __bf16* sVT  = sEDX;

  // ---- stage inputs to LDS as bf16 (incl. irregular neighbor gather) ----
  {
    const float* g = p.edx + (size_t)node * (KNB * EXPD);
    for (int i = tid; i < KNB * EXPD; i += 256) {
      int r = i >> 7, c = i & 127;
      sEDX[r * 136 + c] = f2bf(g[i]);
    }
    const float* ge = p.edge_features + (size_t)node * (KNB * HID);
    for (int i = tid; i < KNB * HID; i += 256) {
      int r = i >> 8, c = i & 255;
      sEF[r * 264 + c] = f2bf(ge[i]);
    }
    for (int i = tid; i < KNB * HID; i += 256) {
      int r = i >> 8, c = i & 255;
      int idx = p.nlist[node * KNB + r];
      sNF[r * 264 + c] = f2bf(p.node_features[(size_t)idx * HID + c]);
    }
  }
  __syncthreads();

  const v8f z = {0.f, 0.f, 0.f, 0.f, 0.f, 0.f, 0.f, 0.f};

  // ---- GEMM1: ea_pre = silu(edx @ We + b) ; then LN -> sEA ----
  {
    v8f c[2][2] = {{z, z}, {z, z}};
    gemm_32tile(sEDX, 136, p.WeT, 128, 128, n0, lane, c);
    for (int mi = 0; mi < 2; ++mi)
      for (int nj = 0; nj < 2; ++nj) {
        int col = n0 + nj * 16 + nl16;
        float b = p.We_b[col];
#pragma unroll
        for (int j = 0; j < 8; ++j) {
          int row = mi * 16 + hlf * 8 + j;
          float x = c[mi][nj][j] + b;
          x = x / (1.f + __expf(-x));
          sScr[row * 260 + col] = x;
        }
      }
  }
  __syncthreads();
  layernorm_rows(sScr, sEA, p.ln_e_s, p.ln_e_b, sRed, sStats, tid);

  // ---- GEMM2: nh = silu(nf @ Wn + b) -> sNH ----
  {
    v8f c[2][2] = {{z, z}, {z, z}};
    gemm_32tile(sNF, 264, p.WnT, 256, 256, n0, lane, c);
    for (int mi = 0; mi < 2; ++mi)
      for (int nj = 0; nj < 2; ++nj) {
        int col = n0 + nj * 16 + nl16;
        float b = p.Wn_b[col];
#pragma unroll
        for (int j = 0; j < 8; ++j) {
          int row = mi * 16 + hlf * 8 + j;
          float x = c[mi][nj][j] + b;
          sNH[row * 264 + col] = f2bf(x / (1.f + __expf(-x)));
        }
      }
  }
  // ---- GEMM3: eh = silu(ef @ Weh + b) -> sEH ----
  {
    v8f c[2][2] = {{z, z}, {z, z}};
    gemm_32tile(sEF, 264, p.WehT, 256, 256, n0, lane, c);
    for (int mi = 0; mi < 2; ++mi)
      for (int nj = 0; nj < 2; ++nj) {
        int col = n0 + nj * 16 + nl16;
        float b = p.Weh_b[col];
#pragma unroll
        for (int j = 0; j < 8; ++j) {
          int row = mi * 16 + hlf * 8 + j;
          float x = c[mi][nj][j] + b;
          sEH[row * 264 + col] = f2bf(x / (1.f + __expf(-x)));
        }
      }
  }
  __syncthreads();

  // ---- GEMM4+5: m=[ea|nh] (K=512); u=sigmoid(m@Wg+b), c=tanh(m@Wc+b);
  //      pre = u*c + (1-u)*eh ; then LN -> sMSG ----
  {
    v8f cu[2][2] = {{z, z}, {z, z}};
    v8f cc[2][2] = {{z, z}, {z, z}};
    for (int k0 = 0; k0 < 512; k0 += 32) {
      const __bf16* A = (k0 < 256) ? sEA : sNH;
      int kk = k0 & 255;
      v16bf a0  = load_A_frag(A, 0,  kk, 264, lane);
      v16bf a1  = load_A_frag(A, 16, kk, 264, lane);
      v16bf bg0 = load_B_frag(p.WgT, n0,      k0, 512, lane);
      v16bf bg1 = load_B_frag(p.WgT, n0 + 16, k0, 512, lane);
      v16bf bc0 = load_B_frag(p.WcT, n0,      k0, 512, lane);
      v16bf bc1 = load_B_frag(p.WcT, n0 + 16, k0, 512, lane);
      wmma_bf16(cu[0][0], a0, bg0); wmma_bf16(cu[0][1], a0, bg1);
      wmma_bf16(cu[1][0], a1, bg0); wmma_bf16(cu[1][1], a1, bg1);
      wmma_bf16(cc[0][0], a0, bc0); wmma_bf16(cc[0][1], a0, bc1);
      wmma_bf16(cc[1][0], a1, bc0); wmma_bf16(cc[1][1], a1, bc1);
    }
    for (int mi = 0; mi < 2; ++mi)
      for (int nj = 0; nj < 2; ++nj) {
        int col = n0 + nj * 16 + nl16;
        float gb = p.Wg_b[col];
        float cb = p.Wc_b[col];
#pragma unroll
        for (int j = 0; j < 8; ++j) {
          int row = mi * 16 + hlf * 8 + j;
          float u = 1.f / (1.f + __expf(-(cu[mi][nj][j] + gb)));
          float t = tanhf(cc[mi][nj][j] + cb);
          float eh = bf2f(sEH[row * 264 + col]);
          sScr[row * 260 + col] = u * t + (1.f - u) * eh;
        }
      }
  }
  __syncthreads();
  layernorm_rows(sScr, sMSG, p.ln_m_s, p.ln_m_b, sRed, sStats, tid);

  // ---- QKV projections from sMSG; K row-major, V stored transposed ----
  {
    v8f c[2][2] = {{z, z}, {z, z}};
    gemm_32tile(sMSG, 264, p.WqT, 256, 256, n0, lane, c);
    for (int mi = 0; mi < 2; ++mi)
      for (int nj = 0; nj < 2; ++nj) {
        int col = n0 + nj * 16 + nl16;
        float b = p.Wq_b[col];
#pragma unroll
        for (int j = 0; j < 8; ++j) {
          int row = mi * 16 + hlf * 8 + j;
          sQ[row * 264 + col] = f2bf(c[mi][nj][j] + b);
        }
      }
  }
  {
    v8f c[2][2] = {{z, z}, {z, z}};
    gemm_32tile(sMSG, 264, p.WkT, 256, 256, n0, lane, c);
    for (int mi = 0; mi < 2; ++mi)
      for (int nj = 0; nj < 2; ++nj) {
        int col = n0 + nj * 16 + nl16;
        float b = p.Wk_b[col];
#pragma unroll
        for (int j = 0; j < 8; ++j) {
          int row = mi * 16 + hlf * 8 + j;
          sKk[row * 264 + col] = f2bf(c[mi][nj][j] + b);
        }
      }
  }
  {
    v8f c[2][2] = {{z, z}, {z, z}};
    gemm_32tile(sMSG, 264, p.WvT, 256, 256, n0, lane, c);
    for (int mi = 0; mi < 2; ++mi)
      for (int nj = 0; nj < 2; ++nj) {
        int col = n0 + nj * 16 + nl16;
        float b = p.Wv_b[col];
#pragma unroll
        for (int j = 0; j < 8; ++j) {
          int row = mi * 16 + hlf * 8 + j;
          sVT[(size_t)col * 40 + row] = f2bf(c[mi][nj][j] + b);  // V^T
        }
      }
  }
  __syncthreads();

  // ---- attention: wave == head; scores, biased softmax, o = a @ v ----
  {
    const int h = wave;
    v8f sc[2][2] = {{z, z}, {z, z}};
    {
      v16bf a0 = load_A_frag(sQ, 0,  h * 32, 264, lane);
      v16bf a1 = load_A_frag(sQ, 16, h * 32, 264, lane);
      v16bf b0 = load_B_frag(sKk, 0,  h * 32, 264, lane);
      v16bf b1 = load_B_frag(sKk, 16, h * 32, 264, lane);
      wmma_bf16(sc[0][0], a0, b0); wmma_bf16(sc[0][1], a0, b1);
      wmma_bf16(sc[1][0], a1, b0); wmma_bf16(sc[1][1], a1, b1);
    }
    const float ascal = p.attn_scalar[h];
    const size_t bb = ((size_t)node * NHEAD + h) * (KNB * KNB);
    const float scale = 0.17677669529663687f;  // 1/sqrt(32)
    for (int mi = 0; mi < 2; ++mi)
      for (int nj = 0; nj < 2; ++nj) {
        int colk = nj * 16 + nl16;
#pragma unroll
        for (int j = 0; j < 8; ++j) {
          int row = mi * 16 + hlf * 8 + j;
          size_t bi = bb + (size_t)row * KNB + colk;
          float b = p.attn_mask[bi] + p.angle[bi] * ascal;
          sSC[(h * 32 + row) * 33 + colk] = sc[mi][nj][j] * scale + b;
        }
      }
    // softmax: one row per lane (32 rows x 32 lanes), wave-private LDS
    {
      float* rp = sSC + (h * 32 + lane) * 33;
      float mx = rp[0];
#pragma unroll
      for (int c2 = 1; c2 < 32; ++c2) mx = fmaxf(mx, rp[c2]);
      float ssum = 0.f;
#pragma unroll
      for (int c2 = 0; c2 < 32; ++c2) ssum += __expf(rp[c2] - mx);
      float inv = 1.f / ssum;
      __bf16* ap = sAT + (size_t)(h * 32 + lane) * 40;
#pragma unroll
      for (int c2 = 0; c2 < 32; ++c2) ap[c2] = f2bf(__expf(rp[c2] - mx) * inv);
    }
    v8f o[2][2] = {{z, z}, {z, z}};
    {
      const __bf16* Ah = sAT + (size_t)(h * 32) * 40;
      v16bf a0 = load_A_frag(Ah, 0,  0, 40, lane);
      v16bf a1 = load_A_frag(Ah, 16, 0, 40, lane);
      v16bf b0 = load_B_frag(sVT, h * 32,      0, 40, lane);
      v16bf b1 = load_B_frag(sVT, h * 32 + 16, 0, 40, lane);
      wmma_bf16(o[0][0], a0, b0); wmma_bf16(o[0][1], a0, b1);
      wmma_bf16(o[1][0], a1, b0); wmma_bf16(o[1][1], a1, b1);
    }
    for (int mi = 0; mi < 2; ++mi)
      for (int nj = 0; nj < 2; ++nj) {
        int col = h * 32 + nj * 16 + nl16;
#pragma unroll
        for (int j = 0; j < 8; ++j) {
          int row = mi * 16 + hlf * 8 + j;
          sO[row * 264 + col] = f2bf(o[mi][nj][j]);
        }
      }
  }
  __syncthreads();

  // ---- GEMM9: edge_output = o @ Wo + b -> global + scratch for node reduce ----
  {
    v8f c[2][2] = {{z, z}, {z, z}};
    gemm_32tile(sO, 264, p.WoT, 256, 256, n0, lane, c);
    float* oe = p.outE + (size_t)node * (KNB * HID);
    for (int mi = 0; mi < 2; ++mi)
      for (int nj = 0; nj < 2; ++nj) {
        int col = n0 + nj * 16 + nl16;
        float b = p.Wo_b[col];
#pragma unroll
        for (int j = 0; j < 8; ++j) {
          int row = mi * 16 + hlf * 8 + j;
          float v = c[mi][nj][j] + b;
          oe[row * HID + col] = v;
          sScr[row * 260 + col] = v;
        }
      }
  }
  __syncthreads();

  // ---- masked-mean aggregate over the K neighbors -> node_output ----
  {
    int col = tid;
    float sum = 0.f;
    int cnt = 0;
#pragma unroll
    for (int r = 0; r < KNB; ++r) {
      if (p.nmask[node * KNB + r]) { sum += sScr[r * 260 + col]; ++cnt; }
    }
    p.outN[(size_t)node * HID + col] = sum / fmaxf((float)cnt, 1.f);
  }
}

extern "C" void kernel_launch(void* const* d_in, const int* in_sizes, int n_in,
                              void* d_out, int out_size, void* d_ws, size_t ws_size,
                              hipStream_t stream) {
  (void)in_sizes; (void)n_in; (void)out_size; (void)ws_size;

  const float* node_features = (const float*)d_in[0];
  const float* edge_features = (const float*)d_in[1];
  const float* edx           = (const float*)d_in[2];
  const float* attn_mask     = (const float*)d_in[3];
  const float* angle         = (const float*)d_in[4];
  const float* attn_scalar   = (const float*)d_in[5];
  const float* We_w  = (const float*)d_in[6];  const float* We_b  = (const float*)d_in[7];
  const float* ln_e_s = (const float*)d_in[8]; const float* ln_e_b = (const float*)d_in[9];
  const float* Wn_w  = (const float*)d_in[10]; const float* Wn_b  = (const float*)d_in[11];
  const float* Weh_w = (const float*)d_in[12]; const float* Weh_b = (const float*)d_in[13];
  const float* Wg_w  = (const float*)d_in[14]; const float* Wg_b  = (const float*)d_in[15];
  const float* Wc_w  = (const float*)d_in[16]; const float* Wc_b  = (const float*)d_in[17];
  const float* ln_m_s = (const float*)d_in[18]; const float* ln_m_b = (const float*)d_in[19];
  const float* Wq_w  = (const float*)d_in[20]; const float* Wq_b  = (const float*)d_in[21];
  const float* Wk_w  = (const float*)d_in[22]; const float* Wk_b  = (const float*)d_in[23];
  const float* Wv_w  = (const float*)d_in[24]; const float* Wv_b  = (const float*)d_in[25];
  const float* Wo_w  = (const float*)d_in[26]; const float* Wo_b  = (const float*)d_in[27];
  const int* nlist            = (const int*)d_in[28];
  const unsigned char* nmask  = (const unsigned char*)d_in[29];

  // bf16 transposed weights in workspace (688128 elems = 1.38 MB; resident in L2)
  __bf16* wsb  = (__bf16*)d_ws;
  __bf16* WeT  = wsb;
  __bf16* WnT  = WeT  + 128 * 256;
  __bf16* WehT = WnT  + 256 * 256;
  __bf16* WgT  = WehT + 256 * 256;
  __bf16* WcT  = WgT  + 512 * 256;
  __bf16* WqT  = WcT  + 512 * 256;
  __bf16* WkT  = WqT  + 256 * 256;
  __bf16* WvT  = WkT  + 256 * 256;
  __bf16* WoT  = WvT  + 256 * 256;

  struct CJ { const float* s; __bf16* d; int Kin; int Nout; };
  const CJ jobs[9] = {
    {We_w,  WeT,  128, 256}, {Wn_w,  WnT,  256, 256}, {Weh_w, WehT, 256, 256},
    {Wg_w,  WgT,  512, 256}, {Wc_w,  WcT,  512, 256}, {Wq_w,  WqT,  256, 256},
    {Wk_w,  WkT,  256, 256}, {Wv_w,  WvT,  256, 256}, {Wo_w,  WoT,  256, 256},
  };
  for (int i = 0; i < 9; ++i) {
    int tot = jobs[i].Kin * jobs[i].Nout;
    wconv_kernel<<<(tot + 255) / 256, 256, 0, stream>>>(jobs[i].s, jobs[i].d,
                                                        jobs[i].Kin, jobs[i].Nout);
  }

  Params p;
  p.node_features = node_features;
  p.edge_features = edge_features;
  p.edx = edx;
  p.attn_mask = attn_mask;
  p.angle = angle;
  p.attn_scalar = attn_scalar;
  p.WeT = WeT;  p.WnT = WnT;  p.WehT = WehT;
  p.WgT = WgT;  p.WcT = WcT;
  p.WqT = WqT;  p.WkT = WkT;  p.WvT = WvT;  p.WoT = WoT;
  p.We_b = We_b;  p.Wn_b = Wn_b;  p.Weh_b = Weh_b;
  p.Wg_b = Wg_b;  p.Wc_b = Wc_b;
  p.Wq_b = Wq_b;  p.Wk_b = Wk_b;  p.Wv_b = Wv_b;  p.Wo_b = Wo_b;
  p.ln_e_s = ln_e_s;  p.ln_e_b = ln_e_b;
  p.ln_m_s = ln_m_s;  p.ln_m_b = ln_m_b;
  p.nlist = nlist;
  p.nmask = nmask;
  p.outN = (float*)d_out;
  p.outE = (float*)d_out + (size_t)N_NODES * HID;

  ega_fused_kernel<<<N_NODES, 256, 0, stream>>>(p);
}
